// Attention_24026047053954
// MI455X (gfx1250) — matrix operand order repaired and checked by reading, not compiled, not run on hardware
//
#include <hip/hip_runtime.h>

// Problem constants (from the reference)
#define B_ 32
#define T_ 2048
#define H_ 1024
#define U_ 1024

typedef __attribute__((ext_vector_type(16))) _Float16 v16h;
typedef __attribute__((ext_vector_type(4)))  _Float16 v4h;
typedef __attribute__((ext_vector_type(8)))  float    v8f;

// ---------------------------------------------------------------------------
// Kernel 0: pre-pack W2 (fp32 [H,U]) into f16 in WMMA B-fragment order.
// For K-tile kt (32 k's), N-tile n (16 u's), lane l:
//   lane's 16 values = W2[kt*32 + ((l>=16)?16:0) + j][n*16 + (l&15)], j=0..15
// stored contiguously -> the score kernel loads each fragment as one v16h
// (two global_load_b128) straight from L2.   Total: 2 MB.
// ---------------------------------------------------------------------------
__global__ void __launch_bounds__(256)
packW2_kernel(const float* __restrict__ W2, _Float16* __restrict__ W2h)
{
    const int idx  = blockIdx.x * 256 + threadIdx.x;  // (H/32)*(U/16)*32 = 65536
    const int lane = idx & 31;
    const int n    = (idx >> 5) & 63;
    const int kt   = idx >> 11;
    const int k0   = kt * 32 + ((lane >= 16) ? 16 : 0);
    const int u    = n * 16 + (lane & 15);
    v16h tmp;
#pragma unroll
    for (int j = 0; j < 16; ++j)
        tmp[j] = (_Float16)W2[(size_t)(k0 + j) * U_ + u];
    *(v16h*)(W2h + (size_t)idx * 16) = tmp;
}

// ---------------------------------------------------------------------------
// Kernel 1: q = dec_hidden @ W1 + b1   -> q[B,U]   (tiny, 67 MFLOP)
// ---------------------------------------------------------------------------
__global__ void __launch_bounds__(256)
qproj_kernel(const float* __restrict__ dec, const float* __restrict__ W1,
             const float* __restrict__ b1, float* __restrict__ q)
{
    const int idx = blockIdx.x * 256 + threadIdx.x;   // B*U threads
    const int b = idx >> 10;
    const int u = idx & (U_ - 1);
    float acc = b1[u];
    const float* d = dec + (size_t)b * H_;
    for (int k = 0; k < H_; ++k)
        acc = fmaf(d[k], W1[(size_t)k * U_ + u], acc);   // W1 coalesced over u
    q[idx] = acc;
}

// ---------------------------------------------------------------------------
// Kernel 2: fused  score[b,t] = V . tanh(q_b + enc[b,t,:]@W2 + b2) + bv
// One block = one batch b, one 32-row tile of T (two 16-row WMMA M-tiles).
// 8 waves x 8 N-tiles cover U=1024.  Each B fragment feeds two WMMAs: the
// loop body is 16 v_wmma_f32_16x16x32_f16 against 8 v16h B loads + one
// vectorized LDS A-tile stage (float4 load -> 2x cvt_pk -> ds_store_b64).
// The 256-MB k tensor is never materialized — it is reduced against V
// inside the accumulator fragments.
// ---------------------------------------------------------------------------
__global__ void __launch_bounds__(256)
score_kernel(const float* __restrict__ enc,     // [B,T,H] fp32
             const _Float16* __restrict__ W2h,  // pre-packed B fragments
             const float* __restrict__ b2,      // [U]
             const float* __restrict__ q,       // [B,U]
             const float* __restrict__ V,       // [U]
             const float* __restrict__ bv,      // [1]
             float* __restrict__ scores)        // [B,T]
{
    __shared__ _Float16 sA[32][36];   // 32 T-rows x 32 K (f16), row pad = 8B
    __shared__ float    ssum[32];     // per-row score partial sums

    const int tid  = threadIdx.x;
    const int lane = tid & 31;
    const int wave = tid >> 5;

    const int b  = blockIdx.x >> 6;           // T/32 = 64 tiles per batch
    const int t0 = (blockIdx.x & 63) << 5;

    if (tid < 32) ssum[tid] = 0.0f;

    // 8 N-tile accumulators per wave, for each of the two M-tiles
    v8f c0[8], c1[8];
#pragma unroll
    for (int j = 0; j < 8; ++j)
#pragma unroll
        for (int e = 0; e < 8; ++e) { c0[j][e] = 0.0f; c1[j][e] = 0.0f; }

    // A-fragment k order used here (contiguous, same as W2h pack): lane holds row M=lane&15;
    // VGPR v holds K = ((lane>=16)?16:0) + 2*v, +1
    const int arow = lane & 15;
    const int ahz  = (lane >= 16) ? 8 : 0;

    // staging coordinates: 8 threads per row, 4 contiguous K each
    const int sr = tid >> 3;
    const int sk = (tid & 7) << 2;

    const float* encb = enc + ((size_t)b * T_ + t0) * H_;

    for (int kt = 0; kt < H_ / 32; ++kt) {
        const int kk = kt * 32;
        __syncthreads();
        // Stage A tile: one float4 per thread -> 4 f16 -> one ds_store_b64
        {
            const float4 e =
                *(const float4*)(encb + (size_t)sr * H_ + kk + sk);
            v4h h;
            h[0] = (_Float16)e.x; h[1] = (_Float16)e.y;
            h[2] = (_Float16)e.z; h[3] = (_Float16)e.w;
            *(v4h*)(&sA[sr][sk]) = h;
            if (kk + 32 < H_)   // pull the next K-slice toward the caches
                __builtin_prefetch(encb + (size_t)sr * H_ + kk + 32 + sk, 0, 1);
        }
        __syncthreads();

        // Two A fragments (M-tiles t0.. and t0+16..), shared by all N-tiles
        v16h af0, af1;
#pragma unroll
        for (int v = 0; v < 8; ++v) {
            const int kb = 2 * ahz + 2 * v;
            af0[2 * v]     = sA[arow][kb];
            af0[2 * v + 1] = sA[arow][kb + 1];
            af1[2 * v]     = sA[16 + arow][kb];
            af1[2 * v + 1] = sA[16 + arow][kb + 1];
        }

#pragma unroll
        for (int j = 0; j < 8; ++j) {
            const int n = wave * 8 + j;
            const v16h bf = *(const v16h*)(
                W2h + ((((size_t)kt * 64 + n) * 32 + lane) * 16));
            c0[j] = __builtin_amdgcn_wmma_f32_16x16x32_f16(
                false, af0, false, bf, (short)0, c0[j], false, false);
            c1[j] = __builtin_amdgcn_wmma_f32_16x16x32_f16(
                false, af1, false, bf, (short)0, c1[j], false, false);
        }
    }

    // Epilogue: tanh(k + q + b2) * V, reduced over U inside the wave.
    // C layout: lane holds col N=lane&15; VGPR r holds row M=r+((lane>=16)?8:0)
    float s0[8], s1[8];
#pragma unroll
    for (int r = 0; r < 8; ++r) { s0[r] = 0.0f; s1[r] = 0.0f; }

#pragma unroll
    for (int j = 0; j < 8; ++j) {
        const int u  = ((wave * 8 + j) << 4) + (lane & 15);
        const float qb = q[(size_t)b * U_ + u] + b2[u];
        const float Vv = V[u];
#pragma unroll
        for (int r = 0; r < 8; ++r) {
            s0[r] += tanhf(c0[j][r] + qb) * Vv;
            s1[r] += tanhf(c1[j][r] + qb) * Vv;
        }
    }

    // Reduce across the 16 lanes sharing a row (xor masks stay in the half)
#pragma unroll
    for (int r = 0; r < 8; ++r) {
        float v0 = s0[r], v1 = s1[r];
        v0 += __shfl_xor(v0, 8, 32);  v1 += __shfl_xor(v1, 8, 32);
        v0 += __shfl_xor(v0, 4, 32);  v1 += __shfl_xor(v1, 4, 32);
        v0 += __shfl_xor(v0, 2, 32);  v1 += __shfl_xor(v1, 2, 32);
        v0 += __shfl_xor(v0, 1, 32);  v1 += __shfl_xor(v1, 1, 32);
        if ((lane & 15) == 0) {
            const int row = r + ((lane >> 4) << 3);
            atomicAdd(&ssum[row], v0);        // ds_add_f32
            atomicAdd(&ssum[16 + row], v1);
        }
    }
    __syncthreads();
    if (tid < 32)
        scores[(size_t)b * T_ + t0 + tid] = ssum[tid] + bv[0];
}

// ---------------------------------------------------------------------------
// Kernel 3: softmax over T (one block per batch row)
// ---------------------------------------------------------------------------
__global__ void __launch_bounds__(256)
softmax_kernel(const float* __restrict__ scores, float* __restrict__ attn)
{
    __shared__ float red[256];
    const int b = blockIdx.x, tid = threadIdx.x;
    const float* s = scores + (size_t)b * T_;

    float mx = -3.4e38f;
    for (int t = tid; t < T_; t += 256) mx = fmaxf(mx, s[t]);
    red[tid] = mx;
    __syncthreads();
    for (int st = 128; st > 0; st >>= 1) {
        if (tid < st) red[tid] = fmaxf(red[tid], red[tid + st]);
        __syncthreads();
    }
    mx = red[0];
    __syncthreads();

    float sum = 0.0f;
    for (int t = tid; t < T_; t += 256) sum += __expf(s[t] - mx);
    red[tid] = sum;
    __syncthreads();
    for (int st = 128; st > 0; st >>= 1) {
        if (tid < st) red[tid] += red[tid + st];
        __syncthreads();
    }
    const float inv = 1.0f / red[0];
    for (int t = tid; t < T_; t += 256)
        attn[(size_t)b * T_ + t] = __expf(s[t] - mx) * inv;
}

// ---------------------------------------------------------------------------
// Kernel 4: context[b,h] = sum_t attn[b,t] * enc[b,t,h]   (second enc pass)
// ---------------------------------------------------------------------------
__global__ void __launch_bounds__(256)
context_kernel(const float* __restrict__ enc, const float* __restrict__ attn,
               float* __restrict__ ctx)
{
    __shared__ float sa[256];
    const int b = blockIdx.x, tid = threadIdx.x;
    const float* eb = enc + (size_t)b * T_ * H_;

    float4 acc = make_float4(0.f, 0.f, 0.f, 0.f);
    for (int t0 = 0; t0 < T_; t0 += 256) {
        sa[tid] = attn[(size_t)b * T_ + t0 + tid];
        __syncthreads();
        for (int tt = 0; tt < 256; ++tt) {
            const float a = sa[tt];
            const float4 e =
                *(const float4*)(eb + (size_t)(t0 + tt) * H_ + tid * 4);
            acc.x = fmaf(a, e.x, acc.x);
            acc.y = fmaf(a, e.y, acc.y);
            acc.z = fmaf(a, e.z, acc.z);
            acc.w = fmaf(a, e.w, acc.w);
        }
        __syncthreads();
    }
    *(float4*)(ctx + (size_t)b * H_ + tid * 4) = acc;
}

// ---------------------------------------------------------------------------
// Launch.  out = [context (B*H) | attn (B*T)]
//          ws  = [q (B*U f32) | scores (B*T f32) | W2h (H*U f16, 2 MB)]
// ---------------------------------------------------------------------------
extern "C" void kernel_launch(void* const* d_in, const int* in_sizes, int n_in,
                              void* d_out, int out_size, void* d_ws, size_t ws_size,
                              hipStream_t stream)
{
    const float* dec = (const float*)d_in[0];
    const float* enc = (const float*)d_in[1];
    const float* W1  = (const float*)d_in[2];
    const float* b1  = (const float*)d_in[3];
    const float* W2  = (const float*)d_in[4];
    const float* b2  = (const float*)d_in[5];
    const float* V   = (const float*)d_in[6];
    const float* bv  = (const float*)d_in[7];

    float* out    = (float*)d_out;
    float* ctx    = out;               // [B, H]
    float* attn   = out + B_ * H_;     // [B, T, 1]

    float*     q      = (float*)d_ws;            // [B, U]
    float*     scores = q + B_ * U_;              // [B, T]
    _Float16*  W2h    = (_Float16*)(scores + B_ * T_);  // [H*U] packed f16

    packW2_kernel <<<(H_ / 32) * (U_ / 16) * 32 / 256, 256, 0, stream>>>(W2, W2h);
    qproj_kernel  <<<(B_ * U_) / 256, 256, 0, stream>>>(dec, W1, b1, q);
    score_kernel  <<<B_ * (T_ / 32),  256, 0, stream>>>(enc, W2h, b2, q, V, bv, scores);
    softmax_kernel<<<B_,              256, 0, stream>>>(scores, attn);
    context_kernel<<<B_,              256, 0, stream>>>(enc, attn, ctx);
}
